// Sageblock_28527172780472
// MI455X (gfx1250) — matrix operand order repaired and checked by reading, not compiled, not run on hardware
//
#include <hip/hip_runtime.h>
#include <hip/hip_bf16.h>
#include <math.h>

// SAGEConv block: mean-aggregate + dual linear + LayerNorm + ELU
// N=100000 nodes, E=1000000 edges, D=H=64.

#define D 64
#define H 64
#define HP 68                 // padded LDS row stride (floats): 16B-aligned, bank-conflict-free
#define WAVES_PER_BLOCK 8

typedef __attribute__((ext_vector_type(16))) __bf16 v16bf;
typedef __attribute__((ext_vector_type(8)))  float  v8f;

__device__ __forceinline__ __bf16 f2bf(float f) { return (__bf16)f; }

__device__ __forceinline__ float elu(float v) {
  return v > 0.0f ? v : expm1f(v);
}

// ---------------- kernel 0: zero the workspace (agg + cnt) ----------------
__global__ void zero_ws_kernel(float* __restrict__ p, long long n) {
  long long i = (long long)blockIdx.x * blockDim.x + threadIdx.x;
  if (i < n) p[i] = 0.0f;
}

// ---------------- kernel 1: pack weights to bf16 B-fragment layout --------
// pk layout: [mat(2)][t(4)][kt(2)][lane(32)][i(16)] bf16. A lane's fragment is
// 32 contiguous bytes -> loaded as one v16bf (2x global_load_b128), no cvt.
__global__ void pack_weights_kernel(const float* __restrict__ Wl,
                                    const float* __restrict__ Wr,
                                    __bf16* __restrict__ pk) {
  int tid = blockIdx.x * blockDim.x + threadIdx.x;
  if (tid >= 2 * 4 * 2 * 32 * 16) return;
  const int i    = tid & 15;
  const int lane = (tid >> 4) & 31;
  const int kt   = (tid >> 9) & 1;
  const int t    = (tid >> 10) & 3;
  const int mat  = (tid >> 12) & 1;
  const float* W = mat ? Wr : Wl;
  const int half = lane >> 4;
  const int ng   = t * 16 + (lane & 15);         // output feature (B column)
  const int k    = kt * 32 + (i < 8 ? half * 8 + i : 16 + half * 8 + (i - 8));      // K index per B-frag layout
  pk[tid] = f2bf(W[ng * D + k]);
}

// ---------------- kernel 2: gather + scatter-add (L2-resident atomics) ----
__global__ void scatter_mean_kernel(const float* __restrict__ x,
                                    const long long* __restrict__ src,
                                    const long long* __restrict__ dst,
                                    float* __restrict__ agg,
                                    float* __restrict__ cnt,
                                    int E) {
  long long t = (long long)blockIdx.x * blockDim.x + threadIdx.x;
  long long e = t >> 4;
  int c4 = (int)(t & 15);
  if (e >= (long long)E) return;
  long long s = src[e];
  long long d = dst[e];
  const float4 v = ((const float4*)x)[s * (D / 4) + c4];
  float* a = agg + d * D + (long long)c4 * 4;
  unsafeAtomicAdd(a + 0, v.x);
  unsafeAtomicAdd(a + 1, v.y);
  unsafeAtomicAdd(a + 2, v.z);
  unsafeAtomicAdd(a + 3, v.w);
  if (c4 == 0) unsafeAtomicAdd(cnt + d, 1.0f);
}

// ---------------- kernel 3: fused mean -> WMMA GEMMs -> bias -> LN -> ELU --
__global__ void __launch_bounds__(256)
sage_wmma_kernel(const float* __restrict__ x,
                 const float* __restrict__ agg,
                 const float* __restrict__ cnt,
                 const v16bf* __restrict__ pk,   // packed bf16 weight fragments
                 const float* __restrict__ bias,
                 const float* __restrict__ gamma,
                 const float* __restrict__ beta,
                 float* __restrict__ out, int N) {
  __shared__ float sh[WAVES_PER_BLOCK][16][HP];
  __shared__ float smu[WAVES_PER_BLOCK][16];
  __shared__ float sis[WAVES_PER_BLOCK][16];

  const int lane = threadIdx.x & 31;
  const int wave = threadIdx.x >> 5;
  const int tile = blockIdx.x * WAVES_PER_BLOCK + wave;
  const int row0 = tile * 16;
  if (row0 >= N) return;  // whole-wave uniform exit: EXEC all-ones for WMMA

  const int half = lane >> 4;
  const int mrow = lane & 15;

  // A fragments (16x32 bf16): lane holds row mrow,
  // K = {k0+8h..+7} U {k0+16+8h..+7}.
  int rg = row0 + mrow;
  if (rg > N - 1) rg = N - 1;  // clamp partial tail tile (masked at output)
  const float* xrow = x   + (long long)rg * D;
  const float* arow = agg + (long long)rg * D;
  const float  inv  = 1.0f / fmaxf(cnt[rg], 1.0f);

  v16bf ax[2], am[2];
  #pragma unroll
  for (int kt = 0; kt < 2; ++kt) {
    const int k0 = kt * 32;
    const int g0 = k0 + half * 8;
    const int g1 = k0 + 16 + half * 8;
    #pragma unroll
    for (int i = 0; i < 8; ++i) {
      ax[kt][i]     = f2bf(xrow[g0 + i]);
      ax[kt][8 + i] = f2bf(xrow[g1 + i]);
      am[kt][i]     = f2bf(arow[g0 + i] * inv);
      am[kt][8 + i] = f2bf(arow[g1 + i] * inv);
    }
  }

  // h[16x64] = mean @ Wl^T + x @ Wr^T + b
  const int ncl = lane & 15;
  #pragma unroll
  for (int t = 0; t < 4; ++t) {
    v8f acc = {};
    const int ng = t * 16 + ncl;
    #pragma unroll
    for (int kt = 0; kt < 2; ++kt) {
      // fragment index f = ((mat*4 + t)*2 + kt); lane's slice at pk[f*32+lane]
      v16bf bl = pk[((0 * 4 + t) * 2 + kt) * 32 + lane];
      acc = __builtin_amdgcn_wmma_f32_16x16x32_bf16(
          false, am[kt], false, bl, (short)0, acc, false, false);
      v16bf br = pk[((1 * 4 + t) * 2 + kt) * 32 + lane];
      acc = __builtin_amdgcn_wmma_f32_16x16x32_bf16(
          false, ax[kt], false, br, (short)0, acc, false, false);
    }
    // C/D layout: VGPR r -> row r + 8*half, column ncl. Bias, spill to LDS.
    const float bn = bias[ng];
    #pragma unroll
    for (int r = 0; r < 8; ++r)
      sh[wave][half * 8 + r][ng] = acc[r] + bn;
  }

  // Intra-wave LDS ordering (wave-private slice => no block barrier needed).
  asm volatile("s_wait_dscnt 0" ::: "memory");

  // Phase 1: lanes 0-15 compute per-row LN stats into LDS.
  if (lane < 16) {
    const float4* r4 = (const float4*)sh[wave][lane];
    float s = 0.0f, s2 = 0.0f;
    #pragma unroll
    for (int k = 0; k < H / 4; ++k) {
      float4 v = r4[k];
      s  += v.x + v.y + v.z + v.w;
      s2 += v.x * v.x + v.y * v.y + v.z * v.z + v.w * v.w;
    }
    const float mu  = s * (1.0f / H);
    const float var = s2 * (1.0f / H) - mu * mu;
    smu[wave][lane] = mu;
    sis[wave][lane] = rsqrtf(var + 1e-5f);
  }
  asm volatile("s_wait_dscnt 0" ::: "memory");

  // Phase 2: all 32 lanes emit the 16x64 tile as coalesced float4 stores.
  #pragma unroll
  for (int j = 0; j < 8; ++j) {
    const int idx = j * 32 + lane;   // 0..255 over (row, col4)
    const int row = idx >> 4;
    const int c4  = idx & 15;
    const int m   = row0 + row;
    if (m < N) {
      const float  mu = smu[wave][row];
      const float  is = sis[wave][row];
      const float4 hv = *(const float4*)&sh[wave][row][c4 * 4];
      const float4 g  = ((const float4*)gamma)[c4];
      const float4 be = ((const float4*)beta)[c4];
      float4 o;
      o.x = elu((hv.x - mu) * is * g.x + be.x);
      o.y = elu((hv.y - mu) * is * g.y + be.y);
      o.z = elu((hv.z - mu) * is * g.z + be.z);
      o.w = elu((hv.w - mu) * is * g.w + be.w);
      ((float4*)(out + (long long)m * H))[c4] = o;
    }
  }
}

// ---------------------------------------------------------------------------
extern "C" void kernel_launch(void* const* d_in, const int* in_sizes, int n_in,
                              void* d_out, int out_size, void* d_ws, size_t ws_size,
                              hipStream_t stream) {
  const float*     x     = (const float*)d_in[0];
  const long long* ei    = (const long long*)d_in[1];  // int64 [2, E]
  const float*     Wl    = (const float*)d_in[2];
  const float*     Wr    = (const float*)d_in[3];
  const float*     b     = (const float*)d_in[4];
  const float*     gamma = (const float*)d_in[5];
  const float*     beta  = (const float*)d_in[6];
  float*           out   = (float*)d_out;

  const int N = in_sizes[0] / D;
  const int E = in_sizes[1] / 2;

  // Workspace: agg[N*64] f32 | cnt[N] f32 | packed bf16 weights (32B aligned).
  float* agg = (float*)d_ws;
  float* cnt = agg + (size_t)N * D;
  size_t pk_off = (((size_t)N * D * 4 + (size_t)N * 4) + 31) & ~(size_t)31;
  __bf16* pk = (__bf16*)((char*)d_ws + pk_off);

  {
    long long n = (long long)N * D + N;
    int blocks = (int)((n + 255) / 256);
    zero_ws_kernel<<<blocks, 256, 0, stream>>>(agg, n);
  }
  {
    pack_weights_kernel<<<32, 256, 0, stream>>>(Wl, Wr, pk);
  }
  {
    long long t = (long long)E * 16;  // (edge, 4-channel chunk) threads
    int blocks = (int)((t + 255) / 256);
    scatter_mean_kernel<<<blocks, 256, 0, stream>>>(x, ei, ei + E, agg, cnt, E);
  }
  {
    int tiles  = (N + 15) / 16;
    int blocks = (tiles + WAVES_PER_BLOCK - 1) / WAVES_PER_BLOCK;
    sage_wmma_kernel<<<blocks, 256, 0, stream>>>(x, agg, cnt, (const v16bf*)pk,
                                                 b, gamma, beta, out, N);
  }
}
